// JetBlock_61830349193720
// MI455X (gfx1250) — matrix operand order repaired and checked by reading, not compiled, not run on hardware
//
#include <hip/hip_runtime.h>
#include <hip/hip_bf16.h>
#include <math.h>

// ---------------- problem constants (match reference) ----------------
#define L   4096
#define H   2048
#define NH  16
#define DK  128
#define DV  256
#define TV  (NH * DV)      // 4096
#define KS  4
#define GR  (H / 8)        // 256, generator reduced dim
#define EPS 1e-5f

// ---------------- WMMA types ----------------
typedef __attribute__((ext_vector_type(16))) __bf16 v16bf;
typedef __attribute__((ext_vector_type(8)))  float  v8f;

__device__ __forceinline__ float act_silu(float x) {
    return x / (1.f + __expf(-x));
}

// ---- CDNA5 async global->LDS copy helpers (ASYNCcnt-tracked) ----
__device__ __forceinline__ void async_b128(void* lds_ptr, const void* gptr) {
    unsigned lds_off = (unsigned)(size_t)lds_ptr;
    unsigned long long ga = (unsigned long long)(size_t)gptr;
    asm volatile("global_load_async_to_lds_b128 %0, %1, off"
                 :: "v"(lds_off), "v"(ga) : "memory");
}
__device__ __forceinline__ void async_b32(void* lds_ptr, const void* gptr) {
    unsigned lds_off = (unsigned)(size_t)lds_ptr;
    unsigned long long ga = (unsigned long long)(size_t)gptr;
    asm volatile("global_load_async_to_lds_b32 %0, %1, off"
                 :: "v"(lds_off), "v"(ga) : "memory");
}
__device__ __forceinline__ void wait_async0() {
    asm volatile("s_wait_asynccnt 0x0" ::: "memory");
}
// wait until only the most recent tile's 4 copies remain in flight
__device__ __forceinline__ void wait_async4() {
    asm volatile("s_wait_asynccnt 0x4" ::: "memory");
}

// =====================================================================
// one-time fp32 -> bf16 conversion (n % 8 == 0)
// =====================================================================
__global__ __launch_bounds__(256)
void cvt_kernel(const float* __restrict__ src, __bf16* __restrict__ dst,
                long long n)
{
    long long i = ((long long)blockIdx.x * 256 + threadIdx.x) * 8;
    if (i >= n) return;
    float4 a = *(const float4*)(src + i);
    float4 b = *(const float4*)(src + i + 4);
    union { __bf16 e[8]; uint4 u; } p;
    p.e[0] = (__bf16)a.x; p.e[1] = (__bf16)a.y;
    p.e[2] = (__bf16)a.z; p.e[3] = (__bf16)a.w;
    p.e[4] = (__bf16)b.x; p.e[5] = (__bf16)b.y;
    p.e[6] = (__bf16)b.z; p.e[7] = (__bf16)b.w;
    *(uint4*)(dst + i) = p.u;
}

// =====================================================================
// one-time fp32 transpose + bf16 convert: W[K,N] -> WT[N,K]
// 32x32 LDS tile, coalesced both sides, guarded for N=16 gate weights.
// =====================================================================
__global__ __launch_bounds__(256)
void cvtT_kernel(const float* __restrict__ W, __bf16* __restrict__ WT,
                 int K, int N)
{
    __shared__ float tile[32][33];
    const int k0 = blockIdx.y * 32, n0 = blockIdx.x * 32;
    const int r = threadIdx.x >> 5;   // 0..7
    const int c = threadIdx.x & 31;
#pragma unroll
    for (int i = 0; i < 4; ++i) {
        int k = k0 + r + i * 8, n = n0 + c;
        tile[r + i * 8][c] = (k < K && n < N) ? W[(long long)k * N + n] : 0.f;
    }
    __syncthreads();
#pragma unroll
    for (int i = 0; i < 4; ++i) {
        int n = n0 + r + i * 8, k = k0 + c;
        if (n < N && k < K) WT[(long long)n * K + k] = (__bf16)tile[c][r + i * 8];
    }
}

// =====================================================================
// bf16 WMMA GEMM: C[M,N] = epilogue(A[M,K] @ BT[N,K]^T)
// A, BT pre-converted bf16; B given transposed so BOTH tiles are
// verbatim byte copies -> staged with global_load_async_to_lds_b128,
// DOUBLE-BUFFERED: tile t+1 streams via the async engine while tile t's
// WMMAs execute; s_wait_asynccnt 0x4 + in-order async completion
// guarantees tile t has landed while t+1 stays in flight.
// Requires M % 128 == 0, K % 32 == 0 (all call sites). N arbitrary
// (async source row clamped; epilogue masks OOB columns).
//
// Block tile 128x128, K-tile 32, 256 threads = 8 waves (wave32),
// wave tile 32x64 = 2x4 v_wmma_f32_16x16x32_bf16.
// Tiles stored row-major-in-K, row stride 40 bf16 (80B = 20 banks ->
// conflict-free b128 fragment reads). Fragments are two contiguous
// 16B chunks each (ISA 7.12.2):
//   A frag: K(e) = (e/8)*16 + lh*8 + (e%8)  -> [row][lh*8], [row][16+lh*8]
//   B frag: K(e) = (e/8)*16 + lh*8 + (e%8)  -> [col][lh*8], [col][16+lh*8]
//
// MODE: 0 = identity, 1 = SiLU, 2 = -exp(aux0[n])*softplus(x+aux1[n]),
//       3 = sigmoid
// =====================================================================
#define BM 128
#define BN 128
#define BK 32
#define LDT 40

template <int MODE>
__global__ __launch_bounds__(256)
void gemm_bf16_wmma(const __bf16* __restrict__ A, const __bf16* __restrict__ BT,
                    float* __restrict__ C, int M, int N, int K,
                    const float* __restrict__ aux0, const float* __restrict__ aux1)
{
    __shared__ __align__(16) __bf16 As[2][BM][LDT];
    __shared__ __align__(16) __bf16 Bs[2][BN][LDT];

    const int tid  = threadIdx.x;
    const int lane = tid & 31;
    const int w    = tid >> 5;   // wave id 0..7
    const int wm   = w >> 1;     // 0..3 : 32-row slice
    const int wn   = w & 1;      // 0..1 : 64-col slice
    const int lr   = lane & 15;
    const int lh   = lane >> 4;

    const int m0 = blockIdx.y * BM;
    const int n0 = blockIdx.x * BN;

    // staging: thread -> tile row tid/2, 16 bf16 (2x b128) at (tid&1)*16
    const int sr = tid >> 1;
    const int sq = (tid & 1) * 16;
    const long long arow = (long long)(m0 + sr) * K;
    int bcl = n0 + sr; if (bcl > N - 1) bcl = N - 1;   // clamp (masked later)
    const long long brow = (long long)bcl * K;

    v8f acc[2][4];
#pragma unroll
    for (int mi = 0; mi < 2; ++mi)
#pragma unroll
        for (int ni = 0; ni < 4; ++ni)
            acc[mi][ni] = (v8f){0.f,0.f,0.f,0.f,0.f,0.f,0.f,0.f};

    // DMA-stage one K-tile (4x global_load_async_to_lds_b128 per thread)
    auto stage = [&](int buf, int kt) {
        async_b128(&As[buf][sr][sq],     A  + arow + kt + sq);
        async_b128(&As[buf][sr][sq + 8], A  + arow + kt + sq + 8);
        async_b128(&Bs[buf][sr][sq],     BT + brow + kt + sq);
        async_b128(&Bs[buf][sr][sq + 8], BT + brow + kt + sq + 8);
    };

    const int nk = K / BK;
    stage(0, 0);                               // prologue

    for (int it = 0; it < nk; ++it) {
        const int buf = it & 1;

        // stream next tile into the other buffer, then wait for current.
        // (safe: all reads of buf^1 finished before last iter's barrier)
        if (it + 1 < nk) {
            stage(buf ^ 1, (it + 1) * BK);
            wait_async4();                     // tile 'it' landed, t+1 in flight
        } else {
            wait_async0();
        }
        __syncthreads();

        // ---- fragments: 2x ds_load_b128 each
        union FB { uint4 q[2]; v16bf v; };
        FB aF[2], bF[4];
#pragma unroll
        for (int mi = 0; mi < 2; ++mi) {
            const int row = wm * 32 + mi * 16 + lr;
            aF[mi].q[0] = *(const uint4*)&As[buf][row][lh * 8];
            aF[mi].q[1] = *(const uint4*)&As[buf][row][16 + lh * 8];
        }
#pragma unroll
        for (int ni = 0; ni < 4; ++ni) {
            const int col = wn * 64 + ni * 16 + lr;
            bF[ni].q[0] = *(const uint4*)&Bs[buf][col][lh * 8];
            bF[ni].q[1] = *(const uint4*)&Bs[buf][col][16 + lh * 8];
        }

#pragma unroll
        for (int mi = 0; mi < 2; ++mi)
#pragma unroll
            for (int ni = 0; ni < 4; ++ni)
                acc[mi][ni] = __builtin_amdgcn_wmma_f32_16x16x32_bf16(
                    false, aF[mi].v, false, bF[ni].v,
                    (short)0, acc[mi][ni], false, false);
        __syncthreads();   // reads of 'buf' done; next iter DMA may overwrite
    }

    // ---- epilogue: C/D layout: VGPR r -> M = lh*8 + r, N = lane%16
#pragma unroll
    for (int mi = 0; mi < 2; ++mi)
#pragma unroll
        for (int ni = 0; ni < 4; ++ni)
#pragma unroll
            for (int r = 0; r < 8; ++r) {
                int row = m0 + wm * 32 + mi * 16 + lh * 8 + r;
                int col = n0 + wn * 64 + ni * 16 + lr;
                if (col < N) {
                    float x = acc[mi][ni][r];
                    if (MODE == 1) {
                        x = act_silu(x);
                    } else if (MODE == 2) {
                        float s  = x + aux1[col];
                        float sp = (s > 20.f) ? s : log1pf(__expf(s));
                        x = -__expf(aux0[col]) * sp;
                    } else if (MODE == 3) {
                        x = 1.f / (1.f + __expf(-x));
                    }
                    C[(long long)row * N + col] = x;
                }
            }
}

// =====================================================================
// q/k L2 norm: one block per (token,head) row, 128 threads = DK.
// =====================================================================
__global__ __launch_bounds__(128)
void qk_norm_kernel(float* __restrict__ q, float* __restrict__ k)
{
    __shared__ float red[128];
    const int tid = threadIdx.x;
    const long long base = (long long)blockIdx.x * DK;

    float qv = q[base + tid];
    float kv = k[base + tid];

    red[tid] = qv * qv; __syncthreads();
#pragma unroll
    for (int s = 64; s > 0; s >>= 1) {
        if (tid < s) red[tid] += red[tid + s];
        __syncthreads();
    }
    float qs = rsqrtf(red[0] + 1e-6f) * 0.08838834764831845f; // DK^-0.5
    __syncthreads();

    red[tid] = kv * kv; __syncthreads();
#pragma unroll
    for (int s = 64; s > 0; s >>= 1) {
        if (tid < s) red[tid] += red[tid + s];
        __syncthreads();
    }
    float ksc = rsqrtf(red[0] + 1e-6f);

    q[base + tid] = qv * qs;
    k[base + tid] = kv * ksc;
}

// =====================================================================
// Fused dynamic depthwise causal conv (VALU, gen_w2 L2-resident)
// =====================================================================
__global__ __launch_bounds__(256)
void conv_kernel(const float* __restrict__ v, const float* __restrict__ tact,
                 const float* __restrict__ conv_w, const float* __restrict__ gen_w2,
                 float* __restrict__ vc)
{
    __shared__ float ta[GR];
    const int t = blockIdx.x;
    const int c = blockIdx.y * 256 + threadIdx.x;

    ta[threadIdx.x] = tact[(long long)t * GR + threadIdx.x];
    __syncthreads();

    float4 w = *(const float4*)(conv_w + (long long)c * KS);

    const float4* g2 = (const float4*)(gen_w2 + (long long)c * KS);
    const long long rowstride4 = (long long)(TV * KS) / 4;
    for (int r = 0; r < GR; ++r) {
        float a = ta[r];
        float4 gw = g2[r * rowstride4];
        w.x = fmaf(a, gw.x, w.x);
        w.y = fmaf(a, gw.y, w.y);
        w.z = fmaf(a, gw.z, w.z);
        w.w = fmaf(a, gw.w, w.w);
    }

    float acc = 0.f;
    {
        int t0 = t - 3, t1 = t - 2, t2 = t - 1;
        float v0 = (t0 >= 0) ? v[(long long)t0 * TV + c] : 0.f;
        float v1 = (t1 >= 0) ? v[(long long)t1 * TV + c] : 0.f;
        float v2 = (t2 >= 0) ? v[(long long)t2 * TV + c] : 0.f;
        float v3 = v[(long long)t * TV + c];
        acc = fmaf(w.x, v0, fmaf(w.y, v1, fmaf(w.z, v2, w.w * v3)));
    }
    vc[(long long)t * TV + c] = act_silu(acc);
}

// =====================================================================
// Fused recurrent gated delta rule; k/q staged via async-to-LDS,
// double-buffered so token t+1 streams while token t computes.
// =====================================================================
__device__ __forceinline__ void async_copy_kq(const float* __restrict__ q,
                                              const float* __restrict__ k,
                                              float* lds_dst, int t, int h, int c)
{
    long long rb = (long long)t * NH * DK + (long long)h * DK;
    const float* gp = (c < DK) ? (k + rb + c) : (q + rb + (c - DK));
    async_b32(lds_dst + c, gp);
}

__global__ __launch_bounds__(256)
void scan_kernel(const float* __restrict__ q, const float* __restrict__ k,
                 const float* __restrict__ v, const float* __restrict__ g,
                 const float* __restrict__ beta, float* __restrict__ o)
{
    __shared__ float kq[2][2 * DK];  // ping-pong: k[0..127], q[0..127]
    const int h = blockIdx.x;
    const int c = threadIdx.x;       // dv column

    float S[DK];
#pragma unroll
    for (int i = 0; i < DK; ++i) S[i] = 0.f;

    async_copy_kq(q, k, kq[0], 0, h, c);

    for (int t = 0; t < L; ++t) {
        const int b = t & 1;
        wait_async0();
        __syncthreads();

        if (t + 1 < L) async_copy_kq(q, k, kq[b ^ 1], t + 1, h, c);

        float gt = g[t * NH + h];
        float bt = beta[t * NH + h];
        float vt = v[(long long)t * TV + h * DV + c];

        float decay = __expf(gt);
        float dot = 0.f;
#pragma unroll
        for (int i = 0; i < DK; ++i) {
            S[i] *= decay;
            dot = fmaf(kq[b][i], S[i], dot);
        }
        float u = (vt - dot) * bt;
        float ot = 0.f;
#pragma unroll
        for (int i = 0; i < DK; ++i) {
            S[i] = fmaf(kq[b][i], u, S[i]);
            ot = fmaf(kq[b][DK + i], S[i], ot);
        }
        o[(long long)t * TV + h * DV + c] = ot;

        __syncthreads();
    }
}

// =====================================================================
// Gated RMSNorm: og = o*silu(z); og *= rsqrt(mean+EPS)*norm_w.
// Emits bf16 directly (feeds the output-projection WMMA GEMM).
// =====================================================================
__global__ __launch_bounds__(256)
void gatednorm_kernel(const float* __restrict__ o, const float* __restrict__ z,
                      const float* __restrict__ nw, __bf16* __restrict__ og)
{
    __shared__ float red[256];
    const int tid = threadIdx.x;
    const long long base = (long long)blockIdx.x * DV;

    float ov = o[base + tid];
    float zv = z[base + tid];
    float x  = ov * act_silu(zv);

    red[tid] = x * x; __syncthreads();
#pragma unroll
    for (int s = 128; s > 0; s >>= 1) {
        if (tid < s) red[tid] += red[tid + s];
        __syncthreads();
    }
    float r = rsqrtf(red[0] * (1.f / DV) + EPS);
    og[base + tid] = (__bf16)(x * r * nw[tid]);
}

// =====================================================================
// launcher
// =====================================================================
extern "C" void kernel_launch(void* const* d_in, const int* in_sizes, int n_in,
                              void* d_out, int out_size, void* d_ws, size_t ws_size,
                              hipStream_t stream)
{
    (void)in_sizes; (void)n_in; (void)out_size; (void)ws_size;

    const float* hs      = (const float*)d_in[0];
    const float* Wq      = (const float*)d_in[1];
    const float* Wk      = (const float*)d_in[2];
    const float* Wv      = (const float*)d_in[3];
    const float* Wa      = (const float*)d_in[4];
    const float* Wb      = (const float*)d_in[5];
    const float* Wg      = (const float*)d_in[6];
    const float* Wo      = (const float*)d_in[7];
    const float* A_log   = (const float*)d_in[8];
    const float* dt_bias = (const float*)d_in[9];
    const float* conv_w  = (const float*)d_in[10];
    const float* gen_w1  = (const float*)d_in[11];
    const float* gen_w2  = (const float*)d_in[12];
    const float* norm_w  = (const float*)d_in[13];

    // ---- fp32 workspace (~261 MB)
    float* ws = (float*)d_ws;
    float* qb = ws;                               // L*NH*DK
    float* kb = qb + (size_t)L * NH * DK;         // L*NH*DK
    float* vb = kb + (size_t)L * NH * DK;         // L*TV
    float* vc = vb + (size_t)L * TV;              // L*TV
    float* tb = vc + (size_t)L * TV;              // L*GR
    float* zb = tb + (size_t)L * GR;              // L*TV
    float* gb = zb + (size_t)L * TV;              // L*NH
    float* bb = gb + (size_t)L * NH;              // L*NH
    float* ob = vb;                               // reuse v (dead after conv)

    // ---- bf16 workspace (~113 MB) right after fp32 region (16B aligned)
    __bf16* bw   = (__bf16*)(bb + (size_t)L * NH);
    __bf16* hsb  = bw;                            // L*H
    __bf16* WqT  = hsb + (size_t)L * H;           // (NH*DK)*H
    __bf16* WkT  = WqT + (size_t)NH * DK * H;     // (NH*DK)*H
    __bf16* WvT  = WkT + (size_t)NH * DK * H;     // TV*H
    __bf16* WgT  = WvT + (size_t)TV * H;          // TV*H
    __bf16* WaT  = WgT + (size_t)TV * H;          // NH*H
    __bf16* WbT  = WaT + (size_t)NH * H;          // NH*H
    __bf16* w1T  = WbT + (size_t)NH * H;          // GR*H
    __bf16* WoT  = w1T + (size_t)GR * H;          // H*TV
    __bf16* ogb  = WoT + (size_t)H * TV;          // L*TV

    dim3 blk(256);
    auto grd  = [](int M, int N) { return dim3((N + BN - 1) / BN, (M + BM - 1) / BM); };
    auto grdT = [](int K, int N) { return dim3((N + 31) / 32, (K + 31) / 32); };

    // ---- one-time bf16 conversion / weight transposition
    cvt_kernel <<<(L * (long long)H) / (8 * 256), blk, 0, stream>>>(hs, hsb, (long long)L * H);
    cvtT_kernel<<<grdT(H, NH * DK), blk, 0, stream>>>(Wq, WqT, H, NH * DK);
    cvtT_kernel<<<grdT(H, NH * DK), blk, 0, stream>>>(Wk, WkT, H, NH * DK);
    cvtT_kernel<<<grdT(H, TV),      blk, 0, stream>>>(Wv, WvT, H, TV);
    cvtT_kernel<<<grdT(H, TV),      blk, 0, stream>>>(Wg, WgT, H, TV);
    cvtT_kernel<<<grdT(H, NH),      blk, 0, stream>>>(Wa, WaT, H, NH);
    cvtT_kernel<<<grdT(H, NH),      blk, 0, stream>>>(Wb, WbT, H, NH);
    cvtT_kernel<<<grdT(H, GR),      blk, 0, stream>>>(gen_w1, w1T, H, GR);
    cvtT_kernel<<<grdT(TV, H),      blk, 0, stream>>>(Wo, WoT, TV, H);

    // ---- projections (double-buffered async-DMA staged bf16 WMMA)
    gemm_bf16_wmma<1><<<grd(L, NH * DK), blk, 0, stream>>>(hsb, WqT, qb, L, NH * DK, H, nullptr, nullptr);
    gemm_bf16_wmma<1><<<grd(L, NH * DK), blk, 0, stream>>>(hsb, WkT, kb, L, NH * DK, H, nullptr, nullptr);
    gemm_bf16_wmma<0><<<grd(L, TV),      blk, 0, stream>>>(hsb, WvT, vb, L, TV, H, nullptr, nullptr);
    gemm_bf16_wmma<0><<<grd(L, TV),      blk, 0, stream>>>(hsb, WgT, zb, L, TV, H, nullptr, nullptr);
    gemm_bf16_wmma<1><<<grd(L, GR),      blk, 0, stream>>>(hsb, w1T, tb, L, GR, H, nullptr, nullptr);
    gemm_bf16_wmma<2><<<grd(L, NH),      blk, 0, stream>>>(hsb, WaT, gb, L, NH, H, A_log, dt_bias);
    gemm_bf16_wmma<3><<<grd(L, NH),      blk, 0, stream>>>(hsb, WbT, bb, L, NH, H, nullptr, nullptr);

    // ---- q/k l2norm (+ scale on q)
    qk_norm_kernel<<<L * NH, 128, 0, stream>>>(qb, kb);

    // ---- fused dynamic causal conv + silu
    conv_kernel<<<dim3(L, TV / 256), 256, 0, stream>>>(vb, tb, conv_w, gen_w2, vc);

    // ---- sequential gated delta rule (o reuses v's buffer)
    scan_kernel<<<NH, 256, 0, stream>>>(qb, kb, vc, gb, bb, ob);

    // ---- gated RMSNorm -> bf16
    gatednorm_kernel<<<L * NH, 256, 0, stream>>>(ob, zb, norm_w, ogb);

    // ---- output projection
    gemm_bf16_wmma<0><<<grd(L, H), blk, 0, stream>>>(ogb, WoT, (float*)d_out, L, H, TV, nullptr, nullptr);
}
